// Net_32624571580892
// MI455X (gfx1250) — compile-verified, matrix operand order checked by reading, not run on hardware
//
#include <hip/hip_runtime.h>
#include <hip/hip_bf16.h>

#define N_NODES 50000
#define N_EDGES 400000
#define F_IN    128
#define F_H     128
#define F_OUT   64

typedef __attribute__((ext_vector_type(16))) __bf16 v16bf;
typedef __attribute__((ext_vector_type(8)))  float  v8f;

static __device__ __forceinline__ __bf16 f2bf(float f) {
  return (__bf16)f;                      // native convert (gfx1250 has HW bf16)
}

// ---------------- degree / normalization ----------------
__global__ void k_degree(const int* __restrict__ src, const int* __restrict__ dst,
                         float* __restrict__ degS, float* __restrict__ degD, int E) {
  int e = blockIdx.x * blockDim.x + threadIdx.x;
  if (e < E) {
    unsafeAtomicAdd(&degS[src[e]], 1.0f);
    unsafeAtomicAdd(&degD[dst[e]], 1.0f);
  }
}

__global__ void k_invsqrt(float* __restrict__ a, float* __restrict__ b, int n) {
  int i = blockIdx.x * blockDim.x + threadIdx.x;
  if (i < n) {
    float x = a[i]; a[i] = (x > 0.0f) ? rsqrtf(x) : 0.0f;
    float y = b[i]; b[i] = (y > 0.0f) ? rsqrtf(y) : 0.0f;
  }
}

// ---------------- pack W (f32 row-major [128 x NCOLS]) into WMMA B-fragment order ----------------
// layout: Wp[((colTile*4 + k0i)*32 + lane)*16 + i] = bf16(W[k][col])
//   k   = k0i*32 + (lane>>4)*8 + (i<8 ? i : 8+i)
//   col = colTile*16 + (lane&15)
// so in the GEMM each lane's B fragment is one contiguous 32B load.
template <int NCOLS>
__global__ void k_pack_w(const float* __restrict__ W, __bf16* __restrict__ Wp) {
  int idx = blockIdx.x * blockDim.x + threadIdx.x;
  if (idx >= 128 * NCOLS) return;
  int i    = idx & 15;
  int lane = (idx >> 4) & 31;
  int k0i  = (idx >> 9) & 3;
  int ct   = idx >> 11;
  int k    = k0i * 32 + ((lane >> 4) << 3) + ((i < 8) ? i : (8 + i));
  int col  = ct * 16 + (lane & 15);
  Wp[idx] = f2bf(W[k * NCOLS + col]);
}

// ---------------- edge gather * ns[src] * (mu + sigma*eps) -> atomic scatter-add ----------------
// one wave32 per edge; each lane owns 4 of the 128 features (float4 path, L2-resident)
__global__ void k_aggregate(const float* __restrict__ x, const float* __restrict__ ns,
                            const int* __restrict__ src, const int* __restrict__ dst,
                            const float* __restrict__ mu, const float* __restrict__ sigma,
                            const float* __restrict__ eps,
                            float* __restrict__ agg, int E) {
  int tid  = blockIdx.x * blockDim.x + threadIdx.x;
  int e    = tid >> 5;
  int lane = tid & 31;
  if (e >= E) return;
  int s = src[e], d = dst[e];
  float sc = ns[s];
  float4 m = *(const float4*)(x + (size_t)s * 128 + lane * 4);
  m.x *= sc; m.y *= sc; m.z *= sc; m.w *= sc;
  if (eps != nullptr) {                      // fused a = mu + sigma*eps (never materialized)
    float mue = mu[e], sge = sigma[e];
    float4 ev = *(const float4*)(eps + (size_t)e * 128 + lane * 4);
    m.x *= fmaf(sge, ev.x, mue);
    m.y *= fmaf(sge, ev.y, mue);
    m.z *= fmaf(sge, ev.z, mue);
    m.w *= fmaf(sge, ev.w, mue);
  }
  float* o = agg + (size_t)d * 128 + lane * 4;
  unsafeAtomicAdd(o + 0, m.x);
  unsafeAtomicAdd(o + 1, m.y);
  unsafeAtomicAdd(o + 2, m.z);
  unsafeAtomicAdd(o + 3, m.w);
}

// ---------------- fused (agg * nd) @ W(bf16) + b  [+ReLU]  via v_wmma_f32_16x16x32_bf16 ----------
// each wave owns one 16-col tile and FIVE 16-row tiles (M=80; 50000 = 80*625).
// B fragment is loaded once per k-step (contiguous 32B) and reused across the 5 row tiles,
// giving 5 independent WMMA accumulation chains to keep the XDL pipe busy.
template <int NCOLS>
__global__ void k_gemm_bias_act(const float* __restrict__ A, const float* __restrict__ nd,
                                const __bf16* __restrict__ Wp, const float* __restrict__ bias,
                                float* __restrict__ out, int relu) {
  const int lane    = threadIdx.x & 31;
  const int wave    = threadIdx.x >> 5;
  const int colTile = blockIdx.x * (blockDim.x >> 5) + wave;
  const int rowBase = blockIdx.y * 80;
  const int r       = lane & 15;
  const int half    = lane >> 4;
  const int col     = colTile * 16 + r;

  float ndv[5];
  const float* Arow[5];
#pragma unroll
  for (int t = 0; t < 5; ++t) {
    int row = rowBase + t * 16 + r;          // A frag: both lane halves hold M = lane&15
    ndv[t]  = nd[row];
    Arow[t] = A + (size_t)row * 128;
  }

  v8f acc[5];
#pragma unroll
  for (int t = 0; t < 5; ++t) acc[t] = (v8f){0.f,0.f,0.f,0.f,0.f,0.f,0.f,0.f};

#pragma unroll
  for (int k0i = 0; k0i < 4; ++k0i) {
    // one contiguous 32B load per lane = full B fragment for this k-step
    v16bf bfrag = *(const v16bf*)(Wp + ((size_t)(colTile * 4 + k0i) * 32 + lane) * 16);
    const int kb = k0i * 32 + half * 8;      // lane-half selects K block (ISA 16-bit A layout)
#pragma unroll
    for (int t = 0; t < 5; ++t) {
      float4 a0 = *(const float4*)(Arow[t] + kb);
      float4 a1 = *(const float4*)(Arow[t] + kb + 4);
      float4 a2 = *(const float4*)(Arow[t] + kb + 16);
      float4 a3 = *(const float4*)(Arow[t] + kb + 20);
      const float nv = ndv[t];
      v16bf af;
      af[0]  = f2bf(a0.x * nv); af[1]  = f2bf(a0.y * nv);
      af[2]  = f2bf(a0.z * nv); af[3]  = f2bf(a0.w * nv);
      af[4]  = f2bf(a1.x * nv); af[5]  = f2bf(a1.y * nv);
      af[6]  = f2bf(a1.z * nv); af[7]  = f2bf(a1.w * nv);
      af[8]  = f2bf(a2.x * nv); af[9]  = f2bf(a2.y * nv);
      af[10] = f2bf(a2.z * nv); af[11] = f2bf(a2.w * nv);
      af[12] = f2bf(a3.x * nv); af[13] = f2bf(a3.y * nv);
      af[14] = f2bf(a3.z * nv); af[15] = f2bf(a3.w * nv);
      acc[t] = __builtin_amdgcn_wmma_f32_16x16x32_bf16(false, af, false, bfrag,
                                                       (short)0, acc[t], false, false);
    }
  }

  const float bv = bias[col];
#pragma unroll
  for (int t = 0; t < 5; ++t) {
#pragma unroll
    for (int v = 0; v < 8; ++v) {            // C/D layout: VGPR v -> M = v (+8 for upper half)
      int orow = rowBase + t * 16 + v + half * 8;
      float val = acc[t][v] + bv;
      if (relu) val = fmaxf(val, 0.0f);
      out[(size_t)orow * NCOLS + col] = val;
    }
  }
}

// ---------------- edge posterior MLP: relu(concat(z[s],z[d])) . W  (+NLL reduction) --------------
// one wave32 per edge; lane owns 4 features of each of the two 128-wide halves
__global__ void k_edge_dist(const float* __restrict__ z,
                            const int* __restrict__ src, const int* __restrict__ dst,
                            const float* __restrict__ Wmu, const float* __restrict__ bmu,
                            const float* __restrict__ Wls, const float* __restrict__ bls,
                            float* __restrict__ mu, float* __restrict__ sigma,
                            float vtarget, float* __restrict__ nllAcc, int E) {
  __shared__ float red[8];
  int tid  = blockIdx.x * blockDim.x + threadIdx.x;
  int e    = tid >> 5;
  int lane = tid & 31;
  int wv   = threadIdx.x >> 5;
  float lane0_nll = 0.0f;
  if (e < E) {
    int s = src[e], d = dst[e];
    float4 zs = *(const float4*)(z + (size_t)s * 128 + lane * 4);
    float4 zd = *(const float4*)(z + (size_t)d * 128 + lane * 4);
    zs.x = fmaxf(zs.x, 0.f); zs.y = fmaxf(zs.y, 0.f);
    zs.z = fmaxf(zs.z, 0.f); zs.w = fmaxf(zs.w, 0.f);
    zd.x = fmaxf(zd.x, 0.f); zd.y = fmaxf(zd.y, 0.f);
    zd.z = fmaxf(zd.z, 0.f); zd.w = fmaxf(zd.w, 0.f);
    float4 wms = *(const float4*)(Wmu + lane * 4);
    float4 wmd = *(const float4*)(Wmu + 128 + lane * 4);
    float4 wss = *(const float4*)(Wls + lane * 4);
    float4 wsd = *(const float4*)(Wls + 128 + lane * 4);
    float pm = zs.x * wms.x + zs.y * wms.y + zs.z * wms.z + zs.w * wms.w
             + zd.x * wmd.x + zd.y * wmd.y + zd.z * wmd.z + zd.w * wmd.w;
    float pl = zs.x * wss.x + zs.y * wss.y + zs.z * wss.z + zs.w * wss.w
             + zd.x * wsd.x + zd.y * wsd.y + zd.z * wsd.z + zd.w * wsd.w;
#pragma unroll
    for (int off = 16; off > 0; off >>= 1) {     // wave32 tree reduction
      pm += __shfl_xor(pm, off, 32);
      pl += __shfl_xor(pl, off, 32);
    }
    if (lane == 0) {
      float m_ = pm + bmu[0];
      float sg = expf(pl + bls[0]);
      if (mu != nullptr) { mu[e] = m_; sigma[e] = sg; }
      float dv = vtarget - m_;
      lane0_nll = -logf(sg) - 0.91893853320467274178f - dv * dv / (2.0f * sg * sg);
    }
  }
  if (lane == 0) red[wv] = lane0_nll;
  __syncthreads();
  if (threadIdx.x == 0) {
    float ssum = 0.0f;
    for (int i = 0; i < (int)(blockDim.x >> 5); ++i) ssum += red[i];
    unsafeAtomicAdd(nllAcc, ssum);
  }
}

__global__ void k_finalize(const float* __restrict__ acc, float* __restrict__ o, float invE) {
  o[0] = -(acc[0] * invE) - (acc[1] * invE);
}

// ---------------- host orchestration ----------------
extern "C" void kernel_launch(void* const* d_in, const int* in_sizes, int n_in,
                              void* d_out, int out_size, void* d_ws, size_t ws_size,
                              hipStream_t stream) {
  (void)in_sizes; (void)n_in; (void)out_size; (void)ws_size;
  const float* x     = (const float*)d_in[0];
  const int*   src   = (const int*)d_in[1];
  const int*   dst   = (const int*)d_in[2];
  const int*   nsrc  = (const int*)d_in[3];
  const int*   ndst  = (const int*)d_in[4];
  const float* epsF  = (const float*)d_in[5];
  const float* epsR  = (const float*)d_in[6];
  const float* Wmu   = (const float*)d_in[7];
  const float* bmu   = (const float*)d_in[8];
  const float* Wls   = (const float*)d_in[9];
  const float* bls   = (const float*)d_in[10];
  const float* Wenc0 = (const float*)d_in[11];
  const float* benc0 = (const float*)d_in[12];
  const float* Wenc1 = (const float*)d_in[13];
  const float* benc1 = (const float*)d_in[14];
  const float* Wg0   = (const float*)d_in[15];
  const float* bg0   = (const float*)d_in[16];
  const float* Wg1   = (const float*)d_in[17];
  const float* bg1   = (const float*)d_in[18];
  const float* Wg2   = (const float*)d_in[19];
  const float* bg2   = (const float*)d_in[20];
  float* out = (float*)d_out;

  char* w = (char*)d_ws;
  size_t off = 0;
  auto take = [&](size_t bytes) -> char* {
    char* p = w + off;
    off += (bytes + 255) & ~(size_t)255;
    return p;
  };
  float*  ns     = (float*)take((size_t)N_NODES * 4);
  float*  nd     = (float*)take((size_t)N_NODES * 4);
  __bf16* wbE0   = (__bf16*)take(128 * 128 * 2);
  __bf16* wbE1   = (__bf16*)take(128 * 128 * 2);
  __bf16* wbG0   = (__bf16*)take(128 * 128 * 2);
  __bf16* wbG1   = (__bf16*)take(128 * 128 * 2);
  __bf16* wbG2   = (__bf16*)take(128 * 64 * 2);
  float*  mu     = (float*)take((size_t)N_EDGES * 4);
  float*  sigma  = (float*)take((size_t)N_EDGES * 4);
  float*  nllAcc = (float*)take(2 * 4);
  float*  agg    = (float*)take((size_t)N_NODES * 128 * 4);
  float*  bufA   = (float*)take((size_t)N_NODES * 128 * 4);
  float*  bufB   = (float*)take((size_t)N_NODES * 128 * 4);

  const int E = N_EDGES, Nn = N_NODES;
  const int ebl    = (E + 255) / 256;
  const int nbl    = (Nn + 255) / 256;
  const int wavebl = (E * 32) / 256;               // one wave32 per edge, exact
  const size_t aggBytes = (size_t)Nn * 128 * 4;
  const dim3 g128(2, Nn / 80), g64(1, Nn / 80);    // 625 row blocks of M=80

  // degrees -> symmetric norms
  hipMemsetAsync(ns, 0, (size_t)Nn * 4, stream);
  hipMemsetAsync(nd, 0, (size_t)Nn * 4, stream);
  hipMemsetAsync(nllAcc, 0, 8, stream);
  k_degree<<<ebl, 256, 0, stream>>>(src, dst, ns, nd, E);
  k_invsqrt<<<nbl, 256, 0, stream>>>(ns, nd, Nn);

  // weights -> bf16, pre-swizzled into B-fragment order (once)
  k_pack_w<128><<<64, 256, 0, stream>>>(Wenc0, wbE0);
  k_pack_w<128><<<64, 256, 0, stream>>>(Wenc1, wbE1);
  k_pack_w<128><<<64, 256, 0, stream>>>(Wg0,   wbG0);
  k_pack_w<128><<<64, 256, 0, stream>>>(Wg1,   wbG1);
  k_pack_w<64><<<32, 256, 0, stream>>>(Wg2,    wbG2);

  // encoder layer 0:  bufA = relu(((x*ns) scatter dst)*nd @ Wenc0 + b)
  hipMemsetAsync(agg, 0, aggBytes, stream);
  k_aggregate<<<wavebl, 256, 0, stream>>>(x, ns, src, dst, nullptr, nullptr, nullptr, agg, E);
  k_gemm_bias_act<128><<<g128, 128, 0, stream>>>(agg, nd, wbE0, benc0, bufA, 1);

  // encoder layer 1:  bufB = z
  hipMemsetAsync(agg, 0, aggBytes, stream);
  k_aggregate<<<wavebl, 256, 0, stream>>>(bufA, ns, src, dst, nullptr, nullptr, nullptr, agg, E);
  k_gemm_bias_act<128><<<g128, 128, 0, stream>>>(agg, nd, wbE1, benc1, bufB, 1);

  // edge posterior (pos: store mu/sigma + NLL@v=1 ; neg: NLL@v=0 only)
  k_edge_dist<<<wavebl, 256, 0, stream>>>(bufB, src, dst, Wmu, bmu, Wls, bls,
                                          mu, sigma, 1.0f, nllAcc + 0, E);
  k_edge_dist<<<wavebl, 256, 0, stream>>>(bufB, nsrc, ndst, Wmu, bmu, Wls, bls,
                                          nullptr, nullptr, 0.0f, nllAcc + 1, E);
  k_finalize<<<1, 1, 0, stream>>>(nllAcc, out + (size_t)Nn * F_OUT, 1.0f / (float)E);

  // stochastic GCN stack (a = mu + sigma*eps fused into the gather)
  hipMemsetAsync(agg, 0, aggBytes, stream);
  k_aggregate<<<wavebl, 256, 0, stream>>>(x, ns, src, dst, mu, sigma, epsF, agg, E);
  k_gemm_bias_act<128><<<g128, 128, 0, stream>>>(agg, nd, wbG0, bg0, bufA, 1);

  hipMemsetAsync(agg, 0, aggBytes, stream);
  k_aggregate<<<wavebl, 256, 0, stream>>>(bufA, ns, src, dst, mu, sigma, epsR, agg, E);
  k_gemm_bias_act<128><<<g128, 128, 0, stream>>>(agg, nd, wbG1, bg1, bufB, 1);

  hipMemsetAsync(agg, 0, aggBytes, stream);
  k_aggregate<<<wavebl, 256, 0, stream>>>(bufB, ns, src, dst, mu, sigma, epsR, agg, E);
  k_gemm_bias_act<64><<<g64, 128, 0, stream>>>(agg, nd, wbG2, bg2, out, 0);
}
